// BaselineModel_75256416960594
// MI455X (gfx1250) — matrix operand order repaired and checked by reading, not compiled, not run on hardware
//
#include <hip/hip_runtime.h>
#include <hip/hip_bf16.h>
#include <float.h>

typedef __attribute__((ext_vector_type(16))) _Float16 v16h;
typedef __attribute__((ext_vector_type(8)))  _Float16 v8h;
typedef __attribute__((ext_vector_type(8)))  float    v8f;

#define HW       64
#define N_PATCH  (HW*HW)      // 4096
#define CCH      128
#define M_BANK   50000
#define OUTSZ    512

// ---------------------------------------------------------------------------
// Kernel 1: L2-normalize each patch vector (dim=channel), emit f16 A (4096x128)
// feat layout: (1,128,64,64) -> feat[c*4096 + n], n = h*64+w
// ---------------------------------------------------------------------------
__global__ void normalize_kernel(const float* __restrict__ feat,
                                 _Float16* __restrict__ A16) {
  __shared__ float s[CCH];
  const int n = blockIdx.x;
  const int c = threadIdx.x;
  float x = feat[(size_t)c * N_PATCH + n];
  s[c] = x * x;
  __syncthreads();
  for (int off = CCH / 2; off > 0; off >>= 1) {
    if (c < off) s[c] += s[c + off];
    __syncthreads();
  }
  float nrm = sqrtf(s[0]);
  float d = fmaxf(nrm, 1e-12f);
  A16[(size_t)n * CCH + c] = (_Float16)(x / d);
}

// ---------------------------------------------------------------------------
// Kernel 2: bank -> f16 copy (row-major 50000x128) + per-row sum of squares
// one wave per bank row; each lane handles 4 channels
// ---------------------------------------------------------------------------
__global__ void bank_prep_kernel(const float* __restrict__ bank,
                                 _Float16* __restrict__ B16,
                                 float* __restrict__ b2) {
  const int wave = threadIdx.x >> 5;
  const int lane = threadIdx.x & 31;
  const int row  = blockIdx.x * 8 + wave;
  if (row >= M_BANK) return;
  const float4 v = ((const float4*)(bank + (size_t)row * CCH))[lane];
  float ss = v.x * v.x + v.y * v.y + v.z * v.z + v.w * v.w;
  #pragma unroll
  for (int m = 16; m >= 1; m >>= 1) ss += __shfl_xor(ss, m, 32);
  if (lane == 0) b2[row] = ss;
  _Float16* dst = B16 + (size_t)row * CCH + lane * 4;
  dst[0] = (_Float16)v.x; dst[1] = (_Float16)v.y;
  dst[2] = (_Float16)v.z; dst[3] = (_Float16)v.w;
}

// ---------------------------------------------------------------------------
// Kernel 3: fused WMMA GEMM (A 4096x128 * bank^T 128x50000) + row-min of
// d2 = 1 + ||b||^2 - 2*dot, then sqrt.
// One block = 32 rows (2 row-tiles), 8 waves sweep 128 columns/iteration.
// Each B fragment feeds 2 WMMAs -> 32 FLOP/byte from L2.
// ---------------------------------------------------------------------------
__global__ void __launch_bounds__(256)
gemm_min_kernel(const _Float16* __restrict__ A16,
                const _Float16* __restrict__ B16,
                const float* __restrict__ b2,
                float* __restrict__ patch) {
  __shared__ float smin[8][32];
  const int wave = threadIdx.x >> 5;
  const int lane = threadIdx.x & 31;
  const int l15  = lane & 15;
  const int hi   = lane >> 4;        // 0 or 1
  const int r0   = blockIdx.x * 32;

  // A fragments: per row-tile t, 16x32 f16 tile per kb, kept in VGPRs.
  // lane<16: halves = K{kb*32+0..15}; lane>=16: K{kb*32+16..31}.
  v16h a[2][4];
  #pragma unroll
  for (int t = 0; t < 2; ++t) {
    const int row = r0 + t * 16 + l15;
    const _Float16* arow = A16 + (size_t)row * CCH;
    #pragma unroll
    for (int kb = 0; kb < 4; ++kb) {
      const int koff = kb * 32 + hi * 16;
      union { v16h v; v8h h[2]; } u;
      u.h[0] = *(const v8h*)(arow + koff);
      u.h[1] = *(const v8h*)(arow + koff + 8);
      a[t][kb] = u.v;
    }
  }

  float vmin[2][8];
  #pragma unroll
  for (int t = 0; t < 2; ++t)
    #pragma unroll
    for (int g = 0; g < 8; ++g) vmin[t][g] = FLT_MAX;

  // 50000 = 3125 * 16 exactly -> no tail, EXEC all-1s around WMMA.
  for (int c0 = wave * 16; c0 < M_BANK; c0 += 128) {
    const int col = c0 + l15;
    // B fragment: 32x16 f16, lane = column, 16 K-contiguous halves.
    const _Float16* brow = B16 + (size_t)col * CCH + hi * 16;
    // unconditional prefetch of next tile at a clamped (always valid) address
    {
      const int pcol = min(col + 128, M_BANK - 1);
      __builtin_prefetch(B16 + (size_t)pcol * CCH + hi * 16, 0, 1);
    }

    v8f acc0 = {};
    v8f acc1 = {};
    #pragma unroll
    for (int kb = 0; kb < 4; ++kb) {
      v16h b = *(const v16h*)(brow + kb * 32);
      acc0 = __builtin_amdgcn_wmma_f32_16x16x32_f16(
          false, a[0][kb], false, b, (short)0, acc0, false, false);
      acc1 = __builtin_amdgcn_wmma_f32_16x16x32_f16(
          false, a[1][kb], false, b, (short)0, acc1, false, false);
    }
    const float bb = 1.0f + b2[col];       // ||a||^2 == 1 after normalization
    #pragma unroll
    for (int g = 0; g < 8; ++g) {
      vmin[0][g] = fminf(vmin[0][g], bb - 2.0f * acc0[g]);
      vmin[1][g] = fminf(vmin[1][g], bb - 2.0f * acc1[g]);
    }
  }

  // min across the 16 lanes of each half-wave (columns), rows stay in place
  #pragma unroll
  for (int t = 0; t < 2; ++t) {
    #pragma unroll
    for (int g = 0; g < 8; ++g) {
      float v = vmin[t][g];
      v = fminf(v, __shfl_xor(v, 1, 32));
      v = fminf(v, __shfl_xor(v, 2, 32));
      v = fminf(v, __shfl_xor(v, 4, 32));
      v = fminf(v, __shfl_xor(v, 8, 32));
      vmin[t][g] = v;
    }
  }
  if (l15 == 0) {
    #pragma unroll
    for (int t = 0; t < 2; ++t)
      #pragma unroll
      for (int g = 0; g < 8; ++g)
        smin[wave][t * 16 + hi * 8 + g] = vmin[t][g];
  }
  __syncthreads();
  if (threadIdx.x < 32) {
    float m = FLT_MAX;
    #pragma unroll
    for (int w = 0; w < 8; ++w) m = fminf(m, smin[w][threadIdx.x]);
    patch[r0 + threadIdx.x] = sqrtf(fmaxf(m, 0.0f));
  }
}

// ---------------------------------------------------------------------------
// Kernel 4: bilinear upsample 64x64 -> 512x512 (half-pixel centers) + block max
// ---------------------------------------------------------------------------
__global__ void upsample_kernel(const float* __restrict__ patch,
                                float* __restrict__ outmap,
                                float* __restrict__ bmax) {
  __shared__ float s[256];
  const int idx = blockIdx.x * 256 + threadIdx.x;
  const int oy = idx >> 9;         // /512
  const int ox = idx & (OUTSZ - 1);
  const float scale = (float)HW / (float)OUTSZ;   // 0.125
  float fy = ((float)oy + 0.5f) * scale - 0.5f;
  float fx = ((float)ox + 0.5f) * scale - 0.5f;
  int y0 = (int)floorf(fy); float wy = fy - (float)y0;
  int x0 = (int)floorf(fx); float wx = fx - (float)x0;
  int y1 = y0 + 1, x1 = x0 + 1;
  y0 = min(max(y0, 0), HW - 1); y1 = min(max(y1, 0), HW - 1);
  x0 = min(max(x0, 0), HW - 1); x1 = min(max(x1, 0), HW - 1);
  float v00 = patch[y0 * HW + x0], v01 = patch[y0 * HW + x1];
  float v10 = patch[y1 * HW + x0], v11 = patch[y1 * HW + x1];
  float v = (1.0f - wy) * ((1.0f - wx) * v00 + wx * v01)
          +          wy * ((1.0f - wx) * v10 + wx * v11);
  outmap[idx] = v;
  s[threadIdx.x] = v;
  __syncthreads();
  for (int off = 128; off > 0; off >>= 1) {
    if (threadIdx.x < off) s[threadIdx.x] = fmaxf(s[threadIdx.x], s[threadIdx.x + off]);
    __syncthreads();
  }
  if (threadIdx.x == 0) bmax[blockIdx.x] = s[0];
}

__global__ void final_max_kernel(const float* __restrict__ bmax,
                                 float* __restrict__ score) {
  __shared__ float s[256];
  float m = -FLT_MAX;
  for (int i = threadIdx.x; i < (OUTSZ * OUTSZ) / 256; i += 256)
    m = fmaxf(m, bmax[i]);
  s[threadIdx.x] = m;
  __syncthreads();
  for (int off = 128; off > 0; off >>= 1) {
    if (threadIdx.x < off) s[threadIdx.x] = fmaxf(s[threadIdx.x], s[threadIdx.x + off]);
    __syncthreads();
  }
  if (threadIdx.x == 0) *score = s[0];
}

// ---------------------------------------------------------------------------
// Workspace layout (all offsets 256B-aligned), total ~14.1 MB:
//   A16   @ 0          4096*128*2   = 1,048,576
//   B16   @ 1048576    50000*128*2  = 12,800,000
//   b2    @ 13848576   50000*4      = 200,000 (padded to 200,704)
//   patch @ 14049280   4096*4
//   bmax  @ 14065664   1024*4
// ---------------------------------------------------------------------------
extern "C" void kernel_launch(void* const* d_in, const int* in_sizes, int n_in,
                              void* d_out, int out_size, void* d_ws, size_t ws_size,
                              hipStream_t stream) {
  const float* feat = (const float*)d_in[0];   // (1,128,64,64) f32
  const float* bank = (const float*)d_in[1];   // (50000,128) f32

  char* ws = (char*)d_ws;
  _Float16* A16  = (_Float16*)(ws + 0);
  _Float16* B16  = (_Float16*)(ws + 1048576);
  float*    b2   = (float*)(ws + 13848576);
  float*    patch= (float*)(ws + 14049280);
  float*    bmax = (float*)(ws + 14065664);

  float* outmap = (float*)d_out;               // 512*512 map
  float* score  = outmap + OUTSZ * OUTSZ;      // scalar

  normalize_kernel<<<N_PATCH, CCH, 0, stream>>>(feat, A16);
  bank_prep_kernel<<<(M_BANK + 7) / 8, 256, 0, stream>>>(bank, B16, b2);
  gemm_min_kernel<<<N_PATCH / 32, 256, 0, stream>>>(A16, B16, b2, patch);
  upsample_kernel<<<(OUTSZ * OUTSZ) / 256, 256, 0, stream>>>(patch, outmap, bmax);
  final_max_kernel<<<1, 256, 0, stream>>>(bmax, score);
}
